// SpikingRetrievalCore_49718541418737
// MI455X (gfx1250) — matrix operand order repaired and checked by reading, not compiled, not run on hardware
//
#include <hip/hip_runtime.h>
#include <cstdint>
#include <cstddef>

// ---------------------------------------------------------------------------
// SpikingRetrievalCore for MI455X (gfx1250, wave32, WMMA + async-LDS)
//
// Shapes: B=256, embed=1024, D=64, E=16, H=1024, T=20
// Dominant op: per-step per-expert S_t[256x64] @ W_e[64x1024]  (~10.7 GFLOP)
// Strategy:
//   - spikes are 0/1 -> exact in f16; experts cast to f16; accumulate f32
//     via v_wmma_f32_16x16x32_f16 (K=64 -> 2 WMMA per 16x16 tile).
//   - membrane state v lives in WMMA accumulator VGPRs for all 20 steps
//     (zero HBM traffic for the 16.8MB state).
//   - expert weights pre-packed into the ISA B-fragment VGPR layout, loaded
//     once per wave and reused for all 20 steps (40 WMMAs per fragment pair).
//   - spike tiles pre-packed into the ISA A-fragment layout; wave 0 stages
//     them into LDS with GLOBAL_LOAD_ASYNC_TO_LDS_B128 (ASYNCcnt), double-
//     buffered so the copy for step t+1 overlaps the WMMAs of step t; all 8
//     waves then read via ds_load (kills the 8x redundant VMEM traffic).
//   - gate-weighted expert combine via LDS reduction (8 waves x 2 experts).
// ---------------------------------------------------------------------------

typedef __attribute__((ext_vector_type(16))) _Float16 v16h;
typedef __attribute__((ext_vector_type(8)))  float    v8f;

// async-to-LDS builtin operand type, per hipcc diagnostic:
//   param0: int __attribute__((vector_size(16))) __device__ *   (global, AS1)
//   param1: LDS destination (AS3)
typedef int v4i_vec __attribute__((vector_size(16)));

#if __has_builtin(__builtin_amdgcn_global_load_async_to_lds_b128)
#define HAVE_ASYNC_LDS 1
#else
#define HAVE_ASYNC_LDS 0
#endif

#define T_STEPS 20
#define LEAK 0.05f     // DT/TAU = 0.001/0.02
#define V_THRESH 0.5f

static constexpr int BATCH = 256;
static constexpr int EMBED = 1024;
static constexpr int DEXP  = 64;
static constexpr int NEXP  = 16;
static constexpr int HID   = 1024;

// workspace layout (bytes)
static constexpr size_t OFF_X    = 0;                                   // 256*64 f32
static constexpr size_t OFF_GATE = OFF_X + (size_t)BATCH * DEXP * 4;    // 256*16 f32
static constexpr size_t OFF_SPK  = OFF_GATE + (size_t)BATCH * NEXP * 4; // T*16*2*512 f16
static constexpr size_t OFF_W    = OFF_SPK + (size_t)T_STEPS * 16 * 2 * 512 * 2; // 16*64*2*512 f16

// deterministic splitmix64 -> uniform [0,1)
__device__ __forceinline__ float u01_hash(uint64_t z) {
    z += 0x9E3779B97F4A7C15ull;
    z = (z ^ (z >> 30)) * 0xBF58476D1CE4E5B9ull;
    z = (z ^ (z >> 27)) * 0x94D049BB133111EBull;
    z = z ^ (z >> 31);
    return (float)(z >> 40) * (1.0f / 16777216.0f);
}

// ---- kernel 1: x = query @ W_in  [256,64] fp32 ------------------------------
__global__ void sretr_x_kernel(const float* __restrict__ q,
                               const float* __restrict__ win,
                               float* __restrict__ x) {
    const int b = blockIdx.x;     // 0..255
    const int d = threadIdx.x;    // 0..63
    float acc = 0.f;
    for (int k = 0; k < EMBED; ++k)
        acc += q[(size_t)b * EMBED + k] * win[(size_t)k * DEXP + d];
    x[b * DEXP + d] = acc;
}

// ---- kernel 2: softmax gate  [256,16] fp32 ---------------------------------
__global__ void sretr_gate_kernel(const float* __restrict__ x,
                                  const float* __restrict__ gk,
                                  const float* __restrict__ bias,
                                  const float* __restrict__ temp,
                                  float* __restrict__ gate) {
    const int b = blockIdx.x;     // 0..255
    const int j = threadIdx.x;    // 0..15
    float acc = bias[j];
    for (int d = 0; d < DEXP; ++d)
        acc += x[b * DEXP + d] * gk[d * NEXP + j];
    __shared__ float lg[NEXP];
    lg[j] = acc / temp[0];
    __syncthreads();
    float mx = lg[0];
    for (int i = 1; i < NEXP; ++i) mx = fmaxf(mx, lg[i]);
    float sum = 0.f;
    for (int i = 0; i < NEXP; ++i) sum += expf(lg[i] - mx);
    gate[b * NEXP + j] = expf(lg[j] - mx) / sum;
}

// ---- kernel 3: Bernoulli spikes packed into WMMA A-fragment layout ---------
// A (16-bit, 16x32 MxK) per ISA 7.12.2: lane l<16 -> M=l, K = j<8?j:j+8
//                                       lane l>=16 -> M=l-16, K = 8 + (j<8?j:j+8)
__global__ void sretr_spike_kernel(const float* __restrict__ x,
                                   _Float16* __restrict__ spk) {
    const int idx = blockIdx.x * blockDim.x + threadIdx.x;
    if (idx >= T_STEPS * BATCH * DEXP) return;
    const int d = idx & 63;
    const int b = (idx >> 6) & 255;
    const int t = idx >> 14;
    const float rate = 1.f / (1.f + expf(-x[b * DEXP + d]));
    const float u = u01_hash((uint64_t)idx);
    const float sp = (u < rate) ? 1.f : 0.f;

    const int btile = b >> 4, m = b & 15, c = d >> 5, kk = d & 31;
    int l, j;
    if      (kk <  8) { l = m;      j = kk;      }
    else if (kk < 16) { l = m + 16; j = kk - 8;  }
    else if (kk < 24) { l = m;      j = kk - 8;  }
    else              { l = m + 16; j = kk - 16; }
    const size_t off = ((size_t)((t * 16 + btile) * 2 + c)) * 512 + l * 16 + j;
    spk[off] = (_Float16)sp;
}

// ---- kernel 4: pack experts fp32 -> f16 B-fragment layout ------------------
// B (16-bit, 32x16 KxN): lane l -> N = l&15, K = c*32 + (l<16?0:8) + (j<8?j:j+8)
__global__ void sretr_packw_kernel(const float* __restrict__ ex,
                                   _Float16* __restrict__ wpk) {
    const int idx = blockIdx.x * blockDim.x + threadIdx.x;
    if (idx >= NEXP * 64 * 2 * 512) return;
    const int j  = idx & 15;
    const int l  = (idx >> 4) & 31;
    const int c  = (idx >> 9) & 1;
    const int ht = (idx >> 10) & 63;
    const int e  = idx >> 16;
    const int k  = c * 32 + ((l < 16) ? 0 : 8) + ((j < 8) ? j : j + 8);
    const int n  = ht * 16 + (l & 15);
    wpk[idx] = (_Float16)ex[((size_t)(e * DEXP + k)) * HID + n];
}

// stage one 2KB spike tile (both K-chunks, all 32 lanes) into LDS.
// 128 x 16B chunks; each of wave0's 32 lanes issues 4 async b128 copies.
__device__ __forceinline__ void sretr_stage_tile(const _Float16* __restrict__ spk,
                                                 _Float16* sbuf,
                                                 int btile, int t, int lane) {
    const char* g = (const char*)(spk + ((size_t)(t * 16 + btile) * 2) * 512);
    char* l = (char*)sbuf;
#if HAVE_ASYNC_LDS
    #pragma unroll
    for (int c = 0; c < 4; ++c) {
        const int chunk = c * 32 + lane;
        __builtin_amdgcn_global_load_async_to_lds_b128(
            (__attribute__((address_space(1))) v4i_vec*)(g + chunk * 16),
            (__attribute__((address_space(3))) v4i_vec*)(l + chunk * 16),
            0, 0);
    }
#else
    #pragma unroll
    for (int c = 0; c < 4; ++c) {
        const int chunk = c * 32 + lane;
        *(v4i_vec*)(l + chunk * 16) = *(const v4i_vec*)(g + chunk * 16);
    }
#endif
}

__device__ __forceinline__ void sretr_wait_async() {
#if HAVE_ASYNC_LDS
#if __has_builtin(__builtin_amdgcn_s_wait_asynccnt)
    __builtin_amdgcn_s_wait_asynccnt(0);
#else
    asm volatile("s_wait_asynccnt 0x0" ::: "memory");
#endif
#endif
}

// ---- kernel 5: main LIF + MoE combine --------------------------------------
__global__ __launch_bounds__(256) void sretr_lif_moe_kernel(
        const _Float16* __restrict__ spk,
        const _Float16* __restrict__ wpk,
        const float* __restrict__ gate,
        float* __restrict__ out) {
    const int btile = blockIdx.x;          // 0..15
    const int htile = blockIdx.y;          // 0..63
    const int wave  = threadIdx.x >> 5;    // 0..7
    const int lane  = threadIdx.x & 31;
    const int e0 = wave * 2, e1 = e0 + 1;

    // double-buffered spike tile (2 x 2KB) + expert-combine scratch
    __shared__ __align__(32) _Float16 sbuf[2][1024];
    __shared__ float red[8][16][16];

    // weight fragments: loaded once, reused for all 20 timesteps
    const v16h b00 = *(const v16h*)(wpk + ((size_t)((e0 * 64 + htile) * 2 + 0)) * 512 + lane * 16);
    const v16h b01 = *(const v16h*)(wpk + ((size_t)((e0 * 64 + htile) * 2 + 1)) * 512 + lane * 16);
    const v16h b10 = *(const v16h*)(wpk + ((size_t)((e1 * 64 + htile) * 2 + 0)) * 512 + lane * 16);
    const v16h b11 = *(const v16h*)(wpk + ((size_t)((e1 * 64 + htile) * 2 + 1)) * 512 + lane * 16);

    const v8f zero = {0.f, 0.f, 0.f, 0.f, 0.f, 0.f, 0.f, 0.f};
    v8f v0 = zero, v1 = zero;     // membrane state, lives in VGPRs across T
    v8f s0 = zero, s1 = zero;     // spike counts

    // prologue: stage t=0 tile
    if (wave == 0) {
        sretr_stage_tile(spk, &sbuf[0][0], btile, 0, lane);
        sretr_wait_async();
    }
    __syncthreads();

    for (int t = 0; t < T_STEPS; ++t) {
        const int cur = t & 1;
        // overlap: stage t+1 into the other buffer while computing t
        if (wave == 0) {
            if (t + 1 < T_STEPS) {
                sretr_stage_tile(spk, &sbuf[cur ^ 1][0], btile, t + 1, lane);
            }
            if (t + 2 < T_STEPS) {
                // pull t+2's lines toward L2/L0 ahead of the async engine
                __builtin_prefetch(spk + ((size_t)((t + 2) * 16 + btile) * 2) * 512 + lane * 16, 0, 1);
            }
        }

        const v16h a0 = *(const v16h*)(&sbuf[cur][0]   + lane * 16);
        const v16h a1 = *(const v16h*)(&sbuf[cur][512] + lane * 16);

        // current = S_t @ W_e  (K=64 -> two chained 16x16x32 WMMAs per expert)
        v8f c0 = __builtin_amdgcn_wmma_f32_16x16x32_f16(false, a0, false, b00, (short)0, zero, false, false);
        c0     = __builtin_amdgcn_wmma_f32_16x16x32_f16(false, a1, false, b01, (short)0, c0,   false, false);
        v8f c1 = __builtin_amdgcn_wmma_f32_16x16x32_f16(false, a0, false, b10, (short)0, zero, false, false);
        c1     = __builtin_amdgcn_wmma_f32_16x16x32_f16(false, a1, false, b11, (short)0, c1,   false, false);

        // LIF: v += (I - v)*leak; spike = v>=0.5; count; reset
        #pragma unroll
        for (int i = 0; i < 8; ++i) {
            float nv = v0[i] + (c0[i] - v0[i]) * LEAK;
            float sp = (nv >= V_THRESH) ? 1.f : 0.f;
            s0[i] += sp;
            v0[i] = nv * (1.f - sp);
            nv = v1[i] + (c1[i] - v1[i]) * LEAK;
            sp = (nv >= V_THRESH) ? 1.f : 0.f;
            s1[i] += sp;
            v1[i] = nv * (1.f - sp);
        }

        // make t+1's tile visible to everyone before the next iteration
        if (wave == 0) sretr_wait_async();
        __syncthreads();
    }

    // gate-weighted partial for this wave's 2 experts
    const int mBase = (lane < 16) ? 0 : 8;
    const int n = lane & 15;
    #pragma unroll
    for (int r = 0; r < 8; ++r) {
        const int bg = btile * 16 + r + mBase;
        const float g0 = gate[bg * NEXP + e0];
        const float g1 = gate[bg * NEXP + e1];
        red[wave][r + mBase][n] = (s0[r] * g0 + s1[r] * g1) * (1.0f / T_STEPS);
    }
    __syncthreads();

    // combine 8 waves (16 experts) -> final [16x16] output tile
    const int m  = threadIdx.x >> 4;   // 0..15
    const int nn = threadIdx.x & 15;
    float acc = 0.f;
    #pragma unroll
    for (int w = 0; w < 8; ++w) acc += red[w][m][nn];
    out[((size_t)(btile * 16 + m)) * HID + htile * 16 + nn] = acc;
}

// ---------------------------------------------------------------------------
extern "C" void kernel_launch(void* const* d_in, const int* in_sizes, int n_in,
                              void* d_out, int out_size, void* d_ws, size_t ws_size,
                              hipStream_t stream) {
    const float* q    = (const float*)d_in[0];  // [256,1024]
    const float* bias = (const float*)d_in[1];  // [16]
    const float* temp = (const float*)d_in[2];  // [1]
    const float* win  = (const float*)d_in[3];  // [1024,64]
    const float* ex   = (const float*)d_in[4];  // [16,64,1024]
    const float* gk   = (const float*)d_in[5];  // [64,16]
    float* out = (float*)d_out;                 // [256,1024]

    char* ws = (char*)d_ws;
    float*    x    = (float*)(ws + OFF_X);
    float*    gate = (float*)(ws + OFF_GATE);
    _Float16* spk  = (_Float16*)(ws + OFF_SPK);
    _Float16* wpk  = (_Float16*)(ws + OFF_W);

    sretr_x_kernel<<<256, 64, 0, stream>>>(q, win, x);
    sretr_gate_kernel<<<256, 16, 0, stream>>>(x, gk, bias, temp, gate);
    sretr_spike_kernel<<<(T_STEPS * BATCH * DEXP + 255) / 256, 256, 0, stream>>>(x, spk);
    sretr_packw_kernel<<<(NEXP * 64 * 2 * 512) / 256, 256, 0, stream>>>(ex, wpk);

    dim3 grid(BATCH / 16, HID / 16);   // 16 x 64 tiles
    sretr_lif_moe_kernel<<<grid, 256, 0, stream>>>(spk, wpk, gate, out);
}
